// ProdLDA_49125835931771
// MI455X (gfx1250) — matrix operand order repaired and checked by reading, not compiled, not run on hardware
//
#include <hip/hip_runtime.h>
#include <hip/hip_bf16.h>
#include <math.h>

// ---------------- problem constants ----------------
#define NV   50000           // vocab V
#define NH   100             // hidden H
#define HP   128             // H padded to 8*16 (8 waves per GEMM1 block)
#define NK   50              // topics K
#define KP   64              // K padded to 2*32
#define NB   2048            // batch B
#define VP   50048           // V padded to multiple of 64 (for W1T rows)
#define BN_EPS 1e-5f

typedef __attribute__((ext_vector_type(16))) __bf16 v16bf;
typedef __attribute__((ext_vector_type(8)))  __bf16 v8bf;
typedef __attribute__((ext_vector_type(8)))  float  v8f;
typedef __attribute__((ext_vector_type(4)))  float  v4f;
typedef __attribute__((ext_vector_type(4)))  unsigned short v4us;

// ---------------- helpers ----------------
__device__ __forceinline__ unsigned short f32_to_bf16(float f) {
    unsigned int u = __float_as_uint(f);
    unsigned int r = (u + 0x7FFFu + ((u >> 16) & 1u)) >> 16;   // RNE
    return (unsigned short)r;
}

__device__ __forceinline__ float softplus_f(float x) {
    return (x > 20.f) ? x : log1pf(expf(x));
}

__device__ __forceinline__ unsigned int next_rand(unsigned int& s) {
    s = s * 1664525u + 1013904223u;
    unsigned int x = s;
    x ^= x >> 17; x *= 0xED5AD4BBu; x ^= x >> 11;
    return x;
}
__device__ __forceinline__ float unif01(unsigned int& s) {  // (0,1]
    return ((next_rand(s) >> 8) + 1u) * (1.0f / 16777217.0f);
}
__device__ __forceinline__ float normal01(unsigned int& s) {
    float u1 = unif01(s), u2 = unif01(s);
    return sqrtf(-2.0f * logf(u1)) * __cosf(6.28318530718f * u2);
}

// Marsaglia–Tsang gamma sampler (with boost for alpha<1)
__device__ float gamma_sample(float alpha, unsigned int& s) {
    float a = alpha, boost = 1.0f;
    if (a < 1.0f) {
        boost = powf(unif01(s), 1.0f / fmaxf(a, 1e-8f));
        a += 1.0f;
    }
    float d = a - (1.0f / 3.0f);
    float c = rsqrtf(9.0f * d);
    float g = d;
    for (int it = 0; it < 32; ++it) {
        float x = normal01(s);
        float t = 1.0f + c * x;
        if (t <= 0.0f) continue;
        float vv = t * t * t;
        float u  = unif01(s);
        if (logf(u) < 0.5f * x * x + d - d * vv + d * logf(vv)) { g = d * vv; break; }
    }
    return g * boost;
}

__device__ float digamma_f(float x) {
    float r = 0.0f;
    while (x < 6.0f) { r -= 1.0f / x; x += 1.0f; }
    float xi = 1.0f / x, xi2 = xi * xi;
    return r + logf(x) - 0.5f * xi
           - xi2 * ((1.0f/12.0f) - xi2 * ((1.0f/120.0f) - xi2 * (1.0f/252.0f)));
}

__device__ __forceinline__ float reduce64(float v, volatile float* sm, int tid) {
    sm[tid] = v; __syncthreads();
    for (int off = 32; off > 0; off >>= 1) {
        if (tid < off) sm[tid] += sm[tid + off];
        __syncthreads();
    }
    float r = sm[0]; __syncthreads();
    return r;
}

// ---------------- K0a: W1 [V,H] f32 -> W1T_bf [HP][VP] bf16 (zero padded) ----
__global__ void k_prep_w1t(const float* __restrict__ W1, unsigned short* __restrict__ W1T) {
    long idx = (long)blockIdx.x * blockDim.x + threadIdx.x;
    long total = (long)HP * VP;
    if (idx >= total) return;
    int n = (int)(idx / VP);
    int v = (int)(idx % VP);
    float val = (n < NH && v < NV) ? W1[(long)v * NH + n] : 0.0f;
    W1T[idx] = f32_to_bf16(val);
}

// ---------------- K0b: Wd [K,V] f32 -> WdT_bf [V][KP] bf16 (zero padded) ----
__global__ void k_prep_wdt(const float* __restrict__ Wd, unsigned short* __restrict__ WdT) {
    long idx = (long)blockIdx.x * blockDim.x + threadIdx.x;
    long total = (long)NV * KP;
    if (idx >= total) return;
    int v = (int)(idx / KP);
    int k = (int)(idx % KP);
    float val = (k < NK) ? Wd[(long)k * NV + v] : 0.0f;
    WdT[idx] = f32_to_bf16(val);
}

// ---------------- K1: h1 = softplus(bows @ W1 + b1), WMMA bf16 --------------
// grid 128 (16-row stripes), block 256 = 8 waves; wave w owns cols [16w,16w+16)
// K-step 64, LDS ping-pong double buffer; one float4 NT load per thread/step.

__device__ __forceinline__ void stage_tile(const float* __restrict__ bows,
                                           unsigned short* __restrict__ dst,
                                           int m0, int k0, int tid) {
    const int r = tid >> 4;          // 0..15 row
    const int q = tid & 15;          // 0..15 float4 within row
    const int kc = k0 + q * 4;
    v4f f = {0.f, 0.f, 0.f, 0.f};
    if (kc + 3 < NV)
        f = __builtin_nontemporal_load((const v4f*)(bows + (long)(m0 + r) * NV + kc));
    v4us u;
    u.x = f32_to_bf16(f.x); u.y = f32_to_bf16(f.y);
    u.z = f32_to_bf16(f.z); u.w = f32_to_bf16(f.w);
    *(v4us*)(dst + r * 64 + q * 4) = u;
}

__global__ __launch_bounds__(256)
void k_gemm1(const float* __restrict__ bows, const unsigned short* __restrict__ W1T,
             const float* __restrict__ b1, float* __restrict__ h1) {
    __shared__ __align__(32) unsigned short aT[2][16 * 64];
    const int m0   = blockIdx.x * 16;
    const int tid  = threadIdx.x;
    const int w    = tid >> 5;
    const int lane = tid & 31;
    const int half = lane >> 4;
    const int nl   = lane & 15;
    const int ncol = w * 16 + nl;                         // 0..127

    v8f acc = {0.f,0.f,0.f,0.f,0.f,0.f,0.f,0.f};
    const unsigned short* w1row = W1T + (long)ncol * VP;

    const int nIter = (NV + 63) / 64;                     // 782
    int p = 0;
    stage_tile(bows, aT[0], m0, 0, tid);
    __syncthreads();

    for (int it = 0; it < nIter; ++it) {
        const int k0 = it * 64;
        if (k0 + 64 < NV) stage_tile(bows, aT[p ^ 1], m0, k0 + 64, tid);
        #pragma unroll
        for (int ks = 0; ks < 2; ++ks) {
            const int kk = ks * 32;
            const __bf16* ab = (const __bf16*)(aT[p] + nl * 64 + kk + half * 16);
            v8bf alo = *(const v8bf*)(ab);
            v8bf ahi = *(const v8bf*)(ab + 8);
            v16bf a = __builtin_shufflevector(alo, ahi,
                        0,1,2,3,4,5,6,7,8,9,10,11,12,13,14,15);
            v16bf b = *(const v16bf*)(const __bf16*)(w1row + k0 + kk + half * 16);
            acc = __builtin_amdgcn_wmma_f32_16x16x32_bf16(false, a, false, b,
                                                          (short)0, acc, false, false);
        }
        __syncthreads();
        p ^= 1;
    }
    // epilogue: bias + softplus, zero padded columns
    #pragma unroll
    for (int r = 0; r < 8; ++r) {
        int m = r + 8 * half;
        float val = (ncol < NH) ? softplus_f(acc[r] + b1[ncol]) : 0.0f;
        h1[(long)(m0 + m) * HP + ncol] = val;
    }
}

// ---------------- K2: h2 = softplus(h1@W2+b2); t = h2@Wt+bt -----------------
__global__ __launch_bounds__(128)
void k_mlp2(const float* __restrict__ h1, const float* __restrict__ W2,
            const float* __restrict__ b2, const float* __restrict__ Wt,
            const float* __restrict__ bt, float* __restrict__ t_out) {
    __shared__ float hrow[HP];
    __shared__ float h2row[HP];
    const int row = blockIdx.x;
    const int tid = threadIdx.x;
    hrow[tid] = h1[(long)row * HP + tid];
    __syncthreads();
    float v = 0.0f;
    if (tid < NH) {
        float acc = b2[tid];
        #pragma unroll 4
        for (int i = 0; i < NH; ++i) acc += hrow[i] * W2[i * NH + tid];
        v = softplus_f(acc);
    }
    h2row[tid] = v;
    __syncthreads();
    if (tid < KP) {
        float o = 0.0f;
        if (tid < NK) {
            float acc = bt[tid];
            #pragma unroll 4
            for (int j = 0; j < NH; ++j) acc += h2row[j] * Wt[j * NK + tid];
            o = acc;
        }
        t_out[(long)row * KP + tid] = o;
    }
}

// ---------------- K3a: batch-norm statistics over batch ---------------------
__global__ __launch_bounds__(256)
void k_bnstats(const float* __restrict__ t, float* __restrict__ mu,
               float* __restrict__ istd) {
    __shared__ float s1[256], s2[256];
    const int k = blockIdx.x, tid = threadIdx.x;
    float a = 0.f, b = 0.f;
    for (int r = tid; r < NB; r += 256) {
        float v = t[(long)r * KP + k];
        a += v; b += v * v;
    }
    s1[tid] = a; s2[tid] = b; __syncthreads();
    for (int off = 128; off > 0; off >>= 1) {
        if (tid < off) { s1[tid] += s1[tid + off]; s2[tid] += s2[tid + off]; }
        __syncthreads();
    }
    if (tid == 0) {
        float m  = s1[0] / (float)NB;
        float vr = s2[0] / (float)NB - m * m;         // biased variance
        mu[k]   = m;
        istd[k] = rsqrtf(vr + BN_EPS);
    }
}

// ---------------- K3b: alpha, Dirichlet rsample, KL --------------------------
__global__ __launch_bounds__(64)
void k_sample_kld(const float* __restrict__ t, const float* __restrict__ mu,
                  const float* __restrict__ istd, const float* __restrict__ gma,
                  const float* __restrict__ beta, float* __restrict__ out_sample,
                  unsigned short* __restrict__ sample_bf, float* __restrict__ out_kld) {
    __shared__ float sm[64];
    const int row = blockIdx.x, k = threadIdx.x;
    float alpha = 0.f, g = 0.f;
    if (k < NK) {
        float tv = t[(long)row * KP + k];
        float th = (tv - mu[k]) * istd[k] * gma[k] + beta[k];
        alpha = expf(th);
        unsigned int st = ((unsigned int)(row * KP + k) * 0x9E3779B9u) ^ 0x85EBCA6Bu;
        st += 42u;
        g = gamma_sample(alpha, st);
    }
    float gs = reduce64(g, sm, k);
    float smp = (k < NK) ? g / gs : 0.0f;
    if (k < NK) out_sample[(long)row * NK + k] = smp;
    sample_bf[(long)row * KP + k] = f32_to_bf16(smp);

    float a0  = reduce64(alpha, sm, k);
    float la  = (k < NK) ? lgammaf(alpha) : 0.0f;
    float sla = reduce64(la, sm, k);
    float dga0 = digamma_f(a0);
    float term = (k < NK) ? (alpha - 1.0f) * (digamma_f(alpha) - dga0) : 0.0f;
    float sterm = reduce64(term, sm, k);
    if (k == 0)
        out_kld[row] = lgammaf(a0) - sla - lgammaf((float)NK) + sterm;
}

// ---------------- K4: decoder GEMM + log_softmax + rec_loss (WMMA) ----------
// grid 128 (16-row stripes), block 512 = 16 waves; K padded to 64 (2 WMMA steps)
__global__ __launch_bounds__(512)
void k_decoder(const unsigned short* __restrict__ sample_bf,
               const unsigned short* __restrict__ WdT,
               const float* __restrict__ bd, const float* __restrict__ bows,
               float* __restrict__ logits, float* __restrict__ rec_loss) {
    __shared__ float red_a[16 * 256];
    __shared__ float red_b[16 * 256];
    __shared__ float lse_sh[16];

    const int m0   = blockIdx.x * 16;
    const int tid  = threadIdx.x;
    const int w    = tid >> 5;
    const int lane = tid & 31;
    const int half = lane >> 4;
    const int nl   = lane & 15;

    // A operand (invariant across the column loop): rows m0..m0+15 of sample_bf
    v16bf av[2];
    #pragma unroll
    for (int ks = 0; ks < 2; ++ks) {
        const __bf16* ap = (const __bf16*)(sample_bf + (long)(m0 + nl) * KP
                                           + ks * 32 + half * 16);
        v8bf lo = *(const v8bf*)(ap);
        v8bf hi = *(const v8bf*)(ap + 8);
        av[ks] = __builtin_shufflevector(lo, hi, 0,1,2,3,4,5,6,7,8,9,10,11,12,13,14,15);
    }

    // pass 1: z = sample@Wd + bd, online per-lane logsumexp state
    float mx[8], se[8];
    #pragma unroll
    for (int r = 0; r < 8; ++r) { mx[r] = -1e30f; se[r] = 0.0f; }

    for (int v0 = w * 16; v0 < NV; v0 += 256) {
        const int v = v0 + nl;
        const __bf16* bp = (const __bf16*)(WdT + (long)v * KP);
        v16bf b0 = *(const v16bf*)(bp + half * 16);
        v16bf b1 = *(const v16bf*)(bp + 32 + half * 16);
        v8f c = {0.f,0.f,0.f,0.f,0.f,0.f,0.f,0.f};
        c = __builtin_amdgcn_wmma_f32_16x16x32_bf16(false, av[0], false, b0,
                                                    (short)0, c, false, false);
        c = __builtin_amdgcn_wmma_f32_16x16x32_bf16(false, av[1], false, b1,
                                                    (short)0, c, false, false);
        const float bdv = bd[v];
        const long base = (long)(m0 + 8 * half) * NV + v;
        #pragma unroll
        for (int r = 0; r < 8; ++r) {
            float z = c[r] + bdv;
            __builtin_nontemporal_store(z, logits + base + (long)r * NV);
            if (z > mx[r]) { se[r] = se[r] * __expf(mx[r] - z) + 1.0f; mx[r] = z; }
            else           { se[r] += __expf(z - mx[r]); }
        }
    }

    // reduce (max, sumexp) partials across lanes/waves -> lse per row
    #pragma unroll
    for (int r = 0; r < 8; ++r) {
        int row  = r + 8 * half;
        int slot = w * 16 + nl;
        red_a[row * 256 + slot] = mx[r];
        red_b[row * 256 + slot] = se[r];
    }
    __syncthreads();
    if (tid < 16) {
        float M = -1e30f, S = 0.0f;
        for (int i = 0; i < 256; ++i) {
            float m = red_a[tid * 256 + i], s = red_b[tid * 256 + i];
            if (m > M) { S = S * __expf(M - m) + s; M = m; }
            else       { S += s * __expf(m - M); }
        }
        lse_sh[tid] = M + logf(S);
    }
    __syncthreads();

    // pass 2: logits = z - lse; rec_loss partial = sum(logits * bows)
    float racc[8];
    #pragma unroll
    for (int r = 0; r < 8; ++r) racc[r] = 0.0f;

    for (int v0 = w * 16; v0 < NV; v0 += 256) {
        const int v = v0 + nl;
        const long base = (long)(m0 + 8 * half) * NV + v;
        __builtin_prefetch(bows + base + 256, 0, 1);
        #pragma unroll
        for (int r = 0; r < 8; ++r) {
            const long ad = base + (long)r * NV;
            float z  = __builtin_nontemporal_load(logits + ad);
            float lg = z - lse_sh[r + 8 * half];
            __builtin_nontemporal_store(lg, logits + ad);
            float bw = __builtin_nontemporal_load(bows + ad);
            racc[r] += lg * bw;
        }
    }
    #pragma unroll
    for (int r = 0; r < 8; ++r)
        red_a[(r + 8 * half) * 256 + w * 16 + nl] = racc[r];
    __syncthreads();
    if (tid < 16) {
        float S = 0.0f;
        for (int i = 0; i < 256; ++i) S += red_a[tid * 256 + i];
        rec_loss[m0 + tid] = -S;
    }
}

// ---------------- host side -------------------------------------------------
static inline size_t align256(size_t x) { return (x + 255) & ~(size_t)255; }

extern "C" void kernel_launch(void* const* d_in, const int* in_sizes, int n_in,
                              void* d_out, int out_size, void* d_ws, size_t ws_size,
                              hipStream_t stream) {
    const float* bows = (const float*)d_in[0];
    const float* W1   = (const float*)d_in[1];
    const float* b1   = (const float*)d_in[2];
    const float* W2   = (const float*)d_in[3];
    const float* b2   = (const float*)d_in[4];
    const float* Wt   = (const float*)d_in[5];
    const float* bt   = (const float*)d_in[6];
    const float* gma  = (const float*)d_in[7];
    const float* beta = (const float*)d_in[8];
    const float* Wd   = (const float*)d_in[9];
    const float* bd   = (const float*)d_in[10];

    float* out        = (float*)d_out;
    float* out_sample = out;                                   // [B*K]
    float* out_logits = out + (size_t)NB * NK;                 // [B*V]
    float* out_kld    = out_logits + (size_t)NB * NV;          // [B]
    float* out_rec    = out_kld + NB;                          // [B]

    // workspace carve-up
    char* ws = (char*)d_ws;
    size_t o = 0;
    unsigned short* W1T = (unsigned short*)(ws + o); o = align256(o + (size_t)HP * VP * 2);
    unsigned short* WdT = (unsigned short*)(ws + o); o = align256(o + (size_t)NV * KP * 2);
    float* h1   = (float*)(ws + o);          o = align256(o + (size_t)NB * HP * 4);
    float* tbuf = (float*)(ws + o);          o = align256(o + (size_t)NB * KP * 4);
    float* mu   = (float*)(ws + o);          o = align256(o + (size_t)KP * 4);
    float* istd = (float*)(ws + o);          o = align256(o + (size_t)KP * 4);
    unsigned short* sample_bf = (unsigned short*)(ws + o);
    o = align256(o + (size_t)NB * KP * 2);
    (void)o; (void)ws_size; (void)n_in; (void)in_sizes; (void)out_size;

    // weight conversion / transpose (bf16, zero-padded)
    {
        long n = (long)HP * VP;
        k_prep_w1t<<<(int)((n + 255) / 256), 256, 0, stream>>>(W1, W1T);
        long m = (long)NV * KP;
        k_prep_wdt<<<(int)((m + 255) / 256), 256, 0, stream>>>(Wd, WdT);
    }
    // encoder GEMM1 (WMMA) + softplus
    k_gemm1<<<NB / 16, 256, 0, stream>>>(bows, W1T, b1, h1);
    // encoder MLP layer 2 + topic projection
    k_mlp2<<<NB, 128, 0, stream>>>(h1, W2, b2, Wt, bt, tbuf);
    // batch-norm statistics
    k_bnstats<<<NK, 256, 0, stream>>>(tbuf, mu, istd);
    // alpha, Dirichlet rsample, KL divergence
    k_sample_kld<<<NB, 64, 0, stream>>>(tbuf, mu, istd, gma, beta,
                                        out_sample, sample_bf, out_kld);
    // decoder GEMM (WMMA) + log_softmax + reconstruction loss
    k_decoder<<<NB / 16, 512, 0, stream>>>(sample_bf, WdT, bd, bows,
                                           out_logits, out_rec);
}
